// GNN_GAMLNET_Model_34832184771010
// MI455X (gfx1250) — matrix operand order repaired and checked by reading, not compiled, not run on hardware
//
#include <hip/hip_runtime.h>

#define N_NODES 50000
#define N_EDGES 800000
#define F_IN    64
#define GIN_F   32
#define H1      128
#define H2      256

typedef __attribute__((ext_vector_type(16))) __bf16    v16bf;
typedef __attribute__((ext_vector_type(8)))  float     v8f;
typedef __attribute__((ext_vector_type(4)))  unsigned  v4u;

static __device__ __forceinline__ unsigned short f2bf(float f) {
    unsigned u = __float_as_uint(f);
    u += 0x7FFFu + ((u >> 16) & 1u);   // round-to-nearest-even
    return (unsigned short)(u >> 16);
}

// ---------------- elementwise / staging kernels ----------------

__global__ void fill_zero_kernel(float* __restrict__ p, int n) {
    int i = blockIdx.x * blockDim.x + threadIdx.x;
    if (i < n) p[i] = 0.0f;
}

__global__ void init_h_kernel(const float* __restrict__ x, float* __restrict__ h, int n) {
    int i = blockIdx.x * blockDim.x + threadIdx.x;
    if (i < n * GIN_F) {
        int r = i >> 5, c = i & 31;
        h[i] = x[r * F_IN + c];
    }
}

__global__ void concat_kernel(const float* __restrict__ x, const float* __restrict__ g,
                              float* __restrict__ hc, int n) {
    int i = blockIdx.x * blockDim.x + threadIdx.x;
    if (i < n * (F_IN + H1)) {
        int r = i / (F_IN + H1), c = i % (F_IN + H1);
        hc[i] = (c < F_IN) ? x[r * F_IN + c] : g[r * H1 + (c - F_IN)];
    }
}

// fp32 -> bf16 activation staging, optional per-row scale (inv_deg for means)
__global__ void to_bf16_kernel(const float* __restrict__ src, unsigned short* __restrict__ dst,
                               const float* __restrict__ rowscale, int rows, int cols) {
    int i = blockIdx.x * blockDim.x + threadIdx.x;
    if (i < rows * cols) {
        float v = src[i];
        if (rowscale) v *= rowscale[i / cols];
        dst[i] = f2bf(v);
    }
}

// weight fp32 K x D row-major  ->  bf16 D x K (transposed) for contiguous B-fragment loads
__global__ void wt_to_bf16_kernel(const float* __restrict__ w, unsigned short* __restrict__ wt,
                                  int K, int D) {
    int i = blockIdx.x * blockDim.x + threadIdx.x;
    if (i < K * D) {
        int k = i / D, o = i % D;
        wt[(size_t)o * K + k] = f2bf(w[i]);
    }
}

// ---------------- graph kernels ----------------

__global__ void deg_kernel(const long long* __restrict__ dst, float* __restrict__ deg, int E) {
    int e = blockIdx.x * blockDim.x + threadIdx.x;
    if (e < E) atomicAdd(&deg[(int)dst[e]], 1.0f);
}

__global__ void invdeg_kernel(float* __restrict__ deg, int n) {
    int i = blockIdx.x * blockDim.x + threadIdx.x;
    if (i < n) deg[i] = 1.0f / fmaxf(deg[i], 1.0f);
}

// out[dst] += h[src], dim multiple of 4; one thread per (edge, 4-float chunk)
__global__ void scatter_add_kernel(const float* __restrict__ h,
                                   const long long* __restrict__ src,
                                   const long long* __restrict__ dst,
                                   float* __restrict__ out, int E, int dim) {
    int chunks = dim >> 2;
    int t = blockIdx.x * blockDim.x + threadIdx.x;
    if (t >= E * chunks) return;
    int e = t / chunks;
    int c = (t - e * chunks) << 2;
    int s = (int)src[e], d = (int)dst[e];
    const float4 v = *(const float4*)(h + (size_t)s * dim + c);
    float* o = out + (size_t)d * dim + c;
    atomicAdd(o + 0, v.x);
    atomicAdd(o + 1, v.y);
    atomicAdd(o + 2, v.z);
    atomicAdd(o + 3, v.w);
}

// ---------------- WMMA GEMM: C[M x Nout] (+)= A[M x K] @ W[K x Nout] ----------------
// A: bf16 row-major. Wt: bf16 pre-transposed (Nout x K). 4 waves per block, each
// wave owns one 16x16 output tile; K looped in steps of 32.
__global__ void __launch_bounds__(128) wmma_gemm_bf16(
    const unsigned short* __restrict__ A,
    const unsigned short* __restrict__ Wt,
    const float* __restrict__ bias,
    float* __restrict__ C,
    int M, int K, int Nout, int accumulate, int relu)
{
    const int wave = threadIdx.x >> 5;
    const int lane = threadIdx.x & 31;
    const int half = lane >> 4;
    const int l16  = lane & 15;
    const int row0 = blockIdx.x * 16;
    const int col0 = (blockIdx.y * 4 + wave) * 16;

    v8f acc = {};
    if (accumulate) {
#pragma unroll
        for (int r = 0; r < 8; ++r)
            acc[r] = C[(size_t)(row0 + 8 * half + r) * Nout + (col0 + l16)];
    }

    const unsigned short* Ap = A  + (size_t)(row0 + l16) * K;  // row M = lane%16
    const unsigned short* Bp = Wt + (size_t)(col0 + l16) * K;  // col N = lane%16

    for (int k0 = 0; k0 < K; k0 += 32) {
        union { v16bf v; v4u u[2]; } a, b;
        // A 16x32 bf16 fragment: e0..7 -> K = half*8 + e ; e8..15 -> K = 16 + half*8 + e
        a.u[0] = *(const v4u*)(Ap + k0 + half * 8);
        a.u[1] = *(const v4u*)(Ap + k0 + 16 + half * 8);
        // B 32x16 bf16 fragment: e0..7 -> K = half*8 + e ; e8..15 -> K = 16 + half*8 + e
        b.u[0] = *(const v4u*)(Bp + k0 + half * 8);
        b.u[1] = *(const v4u*)(Bp + k0 + 16 + half * 8);
        acc = __builtin_amdgcn_wmma_f32_16x16x32_bf16(
                  false, a.v, false, b.v, (short)0, acc, false, false);
    }

#pragma unroll
    for (int r = 0; r < 8; ++r) {
        int row = row0 + 8 * half + r;
        int col = col0 + l16;
        float v = acc[r];
        if (bias) v += bias[col];
        if (relu) v = fmaxf(v, 0.0f);
        C[(size_t)row * Nout + col] = v;
    }
}

// ---------------- final 256 -> 2 head (fp32, wave-per-row reduction) ----------------
__global__ void __launch_bounds__(128) final_layer_kernel(
    const float* __restrict__ h,     // N x 256 (post s0 relu)
    const float* __restrict__ agg2,  // N x 256 (unscaled sum)
    const float* __restrict__ inv,
    const float* __restrict__ wl,    // 256 x 2
    const float* __restrict__ wr,    // 256 x 2
    const float* __restrict__ b,     // 2
    float* __restrict__ out, int n)
{
    int row  = blockIdx.x * 4 + (threadIdx.x >> 5);
    int lane = threadIdx.x & 31;
    if (row >= n) return;
    float s = inv[row];
    float a0 = 0.f, a1 = 0.f;
    const float* hp = h    + (size_t)row * H2;
    const float* mp = agg2 + (size_t)row * H2;
    for (int j = lane; j < H2; j += 32) {
        float hv = hp[j];
        float mv = mp[j] * s;
        a0 += hv * wl[j * 2 + 0] + mv * wr[j * 2 + 0];
        a1 += hv * wl[j * 2 + 1] + mv * wr[j * 2 + 1];
    }
    for (int off = 16; off; off >>= 1) {
        a0 += __shfl_down(a0, off, 32);
        a1 += __shfl_down(a1, off, 32);
    }
    if (lane == 0) {
        out[row * 2 + 0] = a0 + b[0];
        out[row * 2 + 1] = a1 + b[1];
    }
}

// ---------------- host side ----------------

extern "C" void kernel_launch(void* const* d_in, const int* in_sizes, int n_in,
                              void* d_out, int out_size, void* d_ws, size_t ws_size,
                              hipStream_t stream) {
    const int N = N_NODES, E = N_EDGES;

    const float*     x   = (const float*)d_in[0];
    const long long* ei  = (const long long*)d_in[1];
    const long long* srcI = ei;
    const long long* dstI = ei + E;

    const float* W1[3] = {(const float*)d_in[2], (const float*)d_in[6],  (const float*)d_in[10]};
    const float* B1[3] = {(const float*)d_in[3], (const float*)d_in[7],  (const float*)d_in[11]};
    const float* W2[3] = {(const float*)d_in[4], (const float*)d_in[8],  (const float*)d_in[12]};
    const float* B2[3] = {(const float*)d_in[5], (const float*)d_in[9],  (const float*)d_in[13]};
    const float* gout_w = (const float*)d_in[14];
    const float* gout_b = (const float*)d_in[15];
    const float* s0_wl  = (const float*)d_in[16];
    const float* s0_wr  = (const float*)d_in[17];
    const float* s0_b   = (const float*)d_in[18];
    const float* s1_wl  = (const float*)d_in[19];
    const float* s1_wr  = (const float*)d_in[20];
    const float* s1_b   = (const float*)d_in[21];
    float* out = (float*)d_out;

    // workspace carve-up (256B aligned)
    char* ws = (char*)d_ws;
    size_t off = 0;
    auto carve = [&](size_t bytes) -> void* {
        off = (off + 255) & ~(size_t)255;
        void* p = ws + off;
        off += bytes;
        return p;
    };
    float*          inv  = (float*)carve((size_t)N * 4);
    unsigned short* Abf  = (unsigned short*)carve((size_t)N * H2 * 2);
    float*          buf1 = (float*)carve((size_t)N * H2 * 4);
    float*          buf2 = (float*)carve((size_t)N * H2 * 4);
    float*          buf3 = (float*)carve((size_t)N * H2 * 4);
    unsigned short* wbf  = (unsigned short*)carve((size_t)256 * 1024 * 2);

    // bf16 weight staging sub-offsets (element units)
    unsigned short* w1b[3]; unsigned short* w2b[3];
    size_t wo = 0;
    w1b[0] = wbf + wo; wo += (size_t)GIN_F * H1;
    w2b[0] = wbf + wo; wo += (size_t)H1 * H1;
    w1b[1] = wbf + wo; wo += (size_t)H1 * H1;
    w2b[1] = wbf + wo; wo += (size_t)H1 * H1;
    w1b[2] = wbf + wo; wo += (size_t)H1 * H1;
    w2b[2] = wbf + wo; wo += (size_t)H1 * H1;
    unsigned short* goutb = wbf + wo; wo += (size_t)H1 * H1;
    unsigned short* s0lb  = wbf + wo; wo += (size_t)(F_IN + H1) * H2;
    unsigned short* s0rb  = wbf + wo; wo += (size_t)(F_IN + H1) * H2;

    auto nb = [](int n) { return (n + 255) / 256; };

    auto gemm = [&](const unsigned short* A, const unsigned short* Wt, const float* bias,
                    float* C, int K, int Nout, int acc, int relu) {
        dim3 g(N / 16, Nout / 64);
        wmma_gemm_bf16<<<g, 128, 0, stream>>>(A, Wt, bias, C, N, K, Nout, acc, relu);
    };

    // --- degree / inv_deg ---
    fill_zero_kernel<<<nb(N), 256, 0, stream>>>(inv, N);
    deg_kernel<<<nb(E), 256, 0, stream>>>(dstI, inv, E);
    invdeg_kernel<<<nb(N), 256, 0, stream>>>(inv, N);

    // --- weight staging (bf16, transposed) ---
    wt_to_bf16_kernel<<<nb(GIN_F * H1), 256, 0, stream>>>(W1[0], w1b[0], GIN_F, H1);
    wt_to_bf16_kernel<<<nb(H1 * H1), 256, 0, stream>>>(W2[0], w2b[0], H1, H1);
    wt_to_bf16_kernel<<<nb(H1 * H1), 256, 0, stream>>>(W1[1], w1b[1], H1, H1);
    wt_to_bf16_kernel<<<nb(H1 * H1), 256, 0, stream>>>(W2[1], w2b[1], H1, H1);
    wt_to_bf16_kernel<<<nb(H1 * H1), 256, 0, stream>>>(W1[2], w1b[2], H1, H1);
    wt_to_bf16_kernel<<<nb(H1 * H1), 256, 0, stream>>>(W2[2], w2b[2], H1, H1);
    wt_to_bf16_kernel<<<nb(H1 * H1), 256, 0, stream>>>(gout_w, goutb, H1, H1);
    wt_to_bf16_kernel<<<nb((F_IN + H1) * H2), 256, 0, stream>>>(s0_wl, s0lb, F_IN + H1, H2);
    wt_to_bf16_kernel<<<nb((F_IN + H1) * H2), 256, 0, stream>>>(s0_wr, s0rb, F_IN + H1, H2);

    // --- h = x[:, :32]  (buf1) ---
    init_h_kernel<<<nb(N * GIN_F), 256, 0, stream>>>(x, buf1, N);

    // --- GIN layers ---
    int din = GIN_F;
    for (int l = 0; l < 3; ++l) {
        // m = h + segment_sum(h[src], dst)   (buf2)
        hipMemcpyAsync(buf2, buf1, (size_t)N * din * 4, hipMemcpyDeviceToDevice, stream);
        scatter_add_kernel<<<nb(E * (din >> 2)), 256, 0, stream>>>(buf1, srcI, dstI, buf2, E, din);
        // t = relu(m @ w1 + b1)              (buf3)
        to_bf16_kernel<<<nb(N * din), 256, 0, stream>>>(buf2, Abf, nullptr, N, din);
        gemm(Abf, w1b[l], B1[l], buf3, din, H1, 0, 1);
        // h = relu(t @ w2 + b2)              (buf1)
        to_bf16_kernel<<<nb(N * H1), 256, 0, stream>>>(buf3, Abf, nullptr, N, H1);
        gemm(Abf, w2b[l], B2[l], buf1, H1, H1, 0, 1);
        din = H1;
    }

    // --- gin_out = h @ gout_w + gout_b     (buf2) ---
    to_bf16_kernel<<<nb(N * H1), 256, 0, stream>>>(buf1, Abf, nullptr, N, H1);
    gemm(Abf, goutb, gout_b, buf2, H1, H1, 0, 0);

    // --- hcat = [x | gin_out]  (buf3, N x 192) ---
    concat_kernel<<<nb(N * (F_IN + H1)), 256, 0, stream>>>(x, buf2, buf3, N);

    // --- mean1 = segment_sum(hcat) * inv_deg   (sum in buf2; scale folded into bf16 stage) ---
    fill_zero_kernel<<<nb(N * (F_IN + H1)), 256, 0, stream>>>(buf2, N * (F_IN + H1));
    scatter_add_kernel<<<nb(E * ((F_IN + H1) >> 2)), 256, 0, stream>>>(buf3, srcI, dstI, buf2, E, F_IN + H1);

    // --- s0: buf1 = relu(hcat @ wl + mean1 @ wr + b)  (two WMMA passes) ---
    to_bf16_kernel<<<nb(N * (F_IN + H1)), 256, 0, stream>>>(buf3, Abf, nullptr, N, F_IN + H1);
    gemm(Abf, s0lb, s0_b, buf1, F_IN + H1, H2, 0, 0);
    to_bf16_kernel<<<nb(N * (F_IN + H1)), 256, 0, stream>>>(buf2, Abf, inv, N, F_IN + H1);
    gemm(Abf, s0rb, nullptr, buf1, F_IN + H1, H2, 1, 1);

    // --- mean2 sum (buf2, N x 256) ---
    fill_zero_kernel<<<nb(N * H2), 256, 0, stream>>>(buf2, N * H2);
    scatter_add_kernel<<<nb(E * (H2 >> 2)), 256, 0, stream>>>(buf1, srcI, dstI, buf2, E, H2);

    // --- out = h @ s1_wl + mean2 @ s1_wr + b ---
    final_layer_kernel<<<(N + 3) / 4, 128, 0, stream>>>(buf1, buf2, inv, s1_wl, s1_wr, s1_b, out, N);
}
